// Relative_multi_head_attention_51556787421455
// MI455X (gfx1250) — matrix operand order repaired and checked by reading, not compiled, not run on hardware
//
#include <hip/hip_runtime.h>

// ---------------------------------------------------------------------------
// Transformer-XL relative multi-head attention on MI455X (gfx1250).
//
// Roofline: ~71 GFLOP vs ~100 MB essential traffic at 23.3 TB/s -> compute
// bound IF the 512 MB score tensor never touches memory. So: flash-style
// attention (scores live in WMMA accumulators) and every matmul stage on
// v_wmma_f32_16x16x32_bf16 (bf16 in, f32 accumulate; the f32 WMMA path is
// only 16x16x4 = 8x less K per op).
//
// Fragment trick used throughout: in the ISA A-layout (16x32, 16-bit), lane
// L's 16 elements are TWO CONTIGUOUS 8-element runs of row L%16
// (K = 8*hi..+7 and 16+8*hi..+7, hi = L/16). So an A fragment is just two
// 16-byte loads from a row-major tile -> no per-element gathers anywhere.
// B operands are pre-packed once into ISA B-fragment order (lane L: col
// N=L%16, K = (e&7) + 8*hi + 16*(e/8)) so they load as one contiguous 32B/lane chunk.
//
// GEMM software pipeline uses two STATICALLY-NAMED buffer sets with a
// ks-pair loop + peeled tail (no dynamic buffer index), so SROA keeps all
// fragments in VGPRs (a dynamic cur^nxt index makes the compiler demote the
// buffers to scratch -- observed in a previous build).
//
// rel_shift: causal bd[i][j] = qv[i] . r[S-1-(i-j)]; per 16x16 score tile
// this is a 31-wide diagonal band covered by two 16-aligned raw Q.R^T WMMA
// tiles (pt0 = 127-it+jt, pt0+1; pt=128 is a zero pad tile), extracted with
// cross-lane shuffles on the C-layout (element (M,N) <- strip col 15-M+N).
// ---------------------------------------------------------------------------

typedef __attribute__((ext_vector_type(16))) __bf16 v16bf;
typedef __attribute__((ext_vector_type(8)))  float  v8f;

#define SEQ   2048
#define DMOD  512
#define NH    8
#define DH    64
#define BATCH 4
#define ROWS  (BATCH * SEQ)   // 8192

union FragU { uint4 q[2]; v16bf v; };

// ---------------------------------------------------------------------------
// f32 -> bf16 elementwise convert (x, pos_emb only; weight converts are
// fused into the weight pack kernel)
// ---------------------------------------------------------------------------
__global__ __launch_bounds__(256) void cvt_bf16_kernel(const float* __restrict__ src,
                                                       __bf16* __restrict__ dst, int n) {
  int i = blockIdx.x * 256 + threadIdx.x;
  if (i < n) dst[i] = (__bf16)src[i];
}

// ---------------------------------------------------------------------------
// Pack a 512x512 f32 weight matrix into ISA B-fragment order:
// wF[ks(16)][nt(32)][512]; element (K = ks*32 + (e&7) + 8*(lane/16) + 16*(e/8),
// N = nt*16 + lane%16) = W[K][N].  tid IS the storage offset.
// ---------------------------------------------------------------------------
__global__ __launch_bounds__(256) void pack_w_kernel(const float* __restrict__ W,
                                                     __bf16* __restrict__ wF) {
  int tid = blockIdx.x * 256 + threadIdx.x;   // total = 1<<18
  int e    = tid & 15;
  int lane = (tid >> 4) & 31;
  int nt   = (tid >> 9) & 31;
  int ks   = tid >> 14;
  int k = ks * 32 + (e & 7) + 8 * (lane >> 4) + 16 * (e >> 3);
  int n = nt * 16 + (lane & 15);
  wF[tid] = (__bf16)W[(size_t)k * DMOD + n];
}

// ---------------------------------------------------------------------------
// GEMM helpers: load one K-step's fragments / run its 4 WMMAs.
// All indices compile-time constant after inlining -> register resident.
// ---------------------------------------------------------------------------
__device__ __forceinline__ void gemm_load_step(const __bf16* __restrict__ arow,
                                               const __bf16* __restrict__ bbase,
                                               int ks, int hi,
                                               FragU& ua, v16bf (&bF)[4]) {
  ua.q[0] = *(const uint4*)(arow + ks * 32 + 8 * hi);        // K = 8hi..8hi+7
  ua.q[1] = *(const uint4*)(arow + ks * 32 + 16 + 8 * hi);   // K = 16+8hi..+7
  const __bf16* bn = bbase + (size_t)ks * 32 * 512;
#pragma unroll
  for (int t = 0; t < 4; ++t) bF[t] = *(const v16bf*)(bn + t * 512);
}

__device__ __forceinline__ void gemm_mma_step(const FragU& ua, const v16bf (&bF)[4],
                                              v8f (&acc)[4]) {
#pragma unroll
  for (int t = 0; t < 4; ++t)
    acc[t] = __builtin_amdgcn_wmma_f32_16x16x32_bf16(false, ua.v, false, bF[t],
                                                     (short)0, acc[t], false, false);
}

// ---------------------------------------------------------------------------
// GEMM: C[M,512](f32) = A[M,512](bf16 row-major) @ W(pre-packed B-frags).
// One 16x64 tile per wave; ks-pair software pipeline with two named buffer
// sets: the 10 loads of K-step ks+1 are in flight while the 4 WMMAs of
// K-step ks execute.
// ---------------------------------------------------------------------------
__global__ __launch_bounds__(256) void gemm_bf16_kernel(const __bf16* __restrict__ A,
                                                        const __bf16* __restrict__ wF,
                                                        float* __restrict__ C, int M) {
  const int lane = threadIdx.x & 31;
  const int w    = threadIdx.x >> 5;
  const int hi   = lane >> 4;
  const int lo   = lane & 15;

  int tile = blockIdx.x * 8 + w;
  const int tn = tile & 7;                  // 64-wide column group
  const int tm = tile >> 3;
  if (tm >= (M >> 4)) return;               // wave-uniform
  const int r0 = tm << 4;

  const __bf16* arow  = A + (size_t)(r0 + lo) * DMOD;
  const __bf16* bbase = wF + (size_t)(tn * 4) * 512 + (size_t)lane * 16;

  v8f acc[4] = {v8f{}, v8f{}, v8f{}, v8f{}};
  FragU ua0, ua1;
  v16bf b0[4], b1[4];

  gemm_load_step(arow, bbase, 0, hi, ua0, b0);
  for (int ks = 0; ks < 14; ks += 2) {
    gemm_load_step(arow, bbase, ks + 1, hi, ua1, b1);
    __builtin_prefetch(arow + ks * 32 + 256, 0, 1);          // global_prefetch_b8
    gemm_mma_step(ua0, b0, acc);
    gemm_load_step(arow, bbase, ks + 2, hi, ua0, b0);
    gemm_mma_step(ua1, b1, acc);
  }
  gemm_load_step(arow, bbase, 15, hi, ua1, b1);
  gemm_mma_step(ua0, b0, acc);
  gemm_mma_step(ua1, b1, acc);

  // C/D layout: VGPR r -> row r + 8*hi, lane%16 -> col
#pragma unroll
  for (int t = 0; t < 4; ++t)
#pragma unroll
    for (int r = 0; r < 8; ++r)
      C[(size_t)(r0 + r + 8 * hi) * DMOD + tn * 64 + t * 16 + lo] = acc[t][r];
}

// ---------------------------------------------------------------------------
// Attention-operand fragment packers. One thread per packed bf16 element;
// the linear tid IS the storage offset, so the attention kernel reads each
// fragment as one contiguous 32B-per-lane vector load.
// ---------------------------------------------------------------------------

// qu/qv A-frags: [b][h][it(128)][ks(2)][512], bias add fused.
__global__ __launch_bounds__(256) void pack_quqv_kernel(const float* __restrict__ Q,
                                                        const float* __restrict__ ub,
                                                        const float* __restrict__ vb,
                                                        __bf16* __restrict__ quF,
                                                        __bf16* __restrict__ qvF) {
  int tid = blockIdx.x * 256 + threadIdx.x;           // total = 1<<22
  int e    = tid & 15;
  int lane = (tid >> 4) & 31;
  int ks   = (tid >> 9) & 1;
  int it   = (tid >> 10) & 127;
  int h    = (tid >> 17) & 7;
  int b    = tid >> 20;
  int K    = e + 8 * (lane >> 4) + ((e >= 8) ? 8 : 0);        // A-frag K map
  int row  = b * SEQ + it * 16 + (lane & 15);
  int d    = ks * 32 + K;
  float q  = Q[(size_t)row * DMOD + h * DH + d];
  quF[tid] = (__bf16)(q + ub[h * DH + d]);
  qvF[tid] = (__bf16)(q + vb[h * DH + d]);
}

// K^T B-frags for AC: [b][h][jt(128)][ks(2)][512]; element (K=d, N=j) = k[j][d]
__global__ __launch_bounds__(256) void pack_k_kernel(const float* __restrict__ Kf,
                                                     __bf16* __restrict__ ktF) {
  int tid = blockIdx.x * 256 + threadIdx.x;
  int e    = tid & 15;
  int lane = (tid >> 4) & 31;
  int ks   = (tid >> 9) & 1;
  int jt   = (tid >> 10) & 127;
  int h    = (tid >> 17) & 7;
  int b    = tid >> 20;
  int j    = jt * 16 + (lane & 15);
  int d    = ks * 32 + (e & 7) + 8 * (lane >> 4) + 16 * (e >> 3);
  ktF[tid] = (__bf16)Kf[(size_t)(b * SEQ + j) * DMOD + h * DH + d];
}

// V B-frags for P@V: [b][h][js(64)][dt(4)][512]; element (K=j', N=d') = v[j][d]
__global__ __launch_bounds__(256) void pack_v_kernel(const float* __restrict__ Vf,
                                                     __bf16* __restrict__ vF) {
  int tid = blockIdx.x * 256 + threadIdx.x;
  int e    = tid & 15;
  int lane = (tid >> 4) & 31;
  int dt   = (tid >> 9) & 3;
  int js   = (tid >> 11) & 63;
  int h    = (tid >> 17) & 7;
  int b    = tid >> 20;
  int j    = js * 32 + (e & 7) + 8 * (lane >> 4) + 16 * (e >> 3);
  int d    = dt * 16 + (lane & 15);
  vF[tid] = (__bf16)Vf[(size_t)(b * SEQ + j) * DMOD + h * DH + d];
}

// R^T B-frags for BD: [h][pt(129)][ks(2)][512]; pt==128 is a zero pad tile
// for the (masked) p >= S band of the diagonal gather.
__global__ __launch_bounds__(256) void pack_r_kernel(const float* __restrict__ Rf,
                                                     __bf16* __restrict__ rF) {
  int tid = blockIdx.x * 256 + threadIdx.x;           // total = 8*129*2*512
  int e    = tid & 15;
  int lane = (tid >> 4) & 31;
  int ks   = (tid >> 9) & 1;
  int rest = tid >> 10;
  int pt   = rest % 129;
  int h    = rest / 129;
  float val = 0.f;
  if (pt < 128) {
    int p = pt * 16 + (lane & 15);
    int d = ks * 32 + (e & 7) + 8 * (lane >> 4) + 16 * (e >> 3);
    val = Rf[(size_t)p * DMOD + h * DH + d];
  }
  rF[tid] = (__bf16)val;
}

// ---------------------------------------------------------------------------
// Flash attention. One wave per (b, h, 16-row i-block). Online softmax with
// half-wave shuffle reductions (rows of a C-layout tile live in one 16-lane
// group). BD = two extra WMMA tiles + diagonal shuffle gather. P goes through
// a 1 KB row-major LDS tile and is re-read as 2 x ds_load_b128 per lane
// (same-wave DS ops are in-order; single-wave workgroup -> no barrier).
// Output is stored directly as bf16 in the row-major layout the final
// projection GEMM consumes.
// ---------------------------------------------------------------------------
__global__ __launch_bounds__(32) void attn_kernel(const __bf16* __restrict__ quF,
                                                  const __bf16* __restrict__ qvF,
                                                  const __bf16* __restrict__ ktF,
                                                  const __bf16* __restrict__ vF,
                                                  const __bf16* __restrict__ rF,
                                                  __bf16* __restrict__ attnB) {
  const int lane = threadIdx.x;
  const int hi   = lane >> 4;
  const int nI   = lane & 15;
  const int it = blockIdx.x & 127;
  const int h  = (blockIdx.x >> 7) & 7;
  const int b  = blockIdx.x >> 10;
  const int i0 = it * 16;

  alignas(64) __shared__ __bf16 Plds[16][32];

  const size_t quBase = ((size_t)(b * NH + h) * 128 + it) * 2;
  v16bf a_qu[2], a_qv[2];
#pragma unroll
  for (int ks = 0; ks < 2; ++ks) {
    a_qu[ks] = *(const v16bf*)(quF + (quBase + ks) * 512 + (size_t)lane * 16);
    a_qv[ks] = *(const v16bf*)(qvF + (quBase + ks) * 512 + (size_t)lane * 16);
  }

  v8f o0 = {}, o1 = {}, o2 = {}, o3 = {};
  float mrow[8], lrow[8];
#pragma unroll
  for (int r = 0; r < 8; ++r) { mrow[r] = -3.0e38f; lrow[r] = 0.f; }

  const int nsteps = (i0 + 16 + 31) >> 5;      // causal: j < i0+16
  for (int js = 0; js < nsteps; ++js) {
    float s[2][8];
    float pv[2][8];
#pragma unroll
    for (int sub = 0; sub < 2; ++sub) {
      const int j0 = js * 32 + sub * 16;
      if (j0 <= i0 + 15) {                      // wave-uniform (EXEC stays all-1)
        const int jt = j0 >> 4;
        // ---- AC = (q+u) . k^T ----
        const size_t kb = ((size_t)(b * NH + h) * 128 + jt) * 2;
        v8f ac = {};
        ac = __builtin_amdgcn_wmma_f32_16x16x32_bf16(false, a_qu[0], false,
             *(const v16bf*)(ktF + kb * 512 + (size_t)lane * 16), (short)0, ac, false, false);
        ac = __builtin_amdgcn_wmma_f32_16x16x32_bf16(false, a_qu[1], false,
             *(const v16bf*)(ktF + (kb + 1) * 512 + (size_t)lane * 16), (short)0, ac, false, false);
        // ---- BD strip: raw (q+v) . r^T tiles at pt0, pt0+1 ----
        const int pt0 = 127 - it + jt;
        const size_t rbL = ((size_t)h * 129 + pt0) * 2;
        const size_t rbR = ((size_t)h * 129 + pt0 + 1) * 2;
        v8f cl = {}, cr = {};
        cl = __builtin_amdgcn_wmma_f32_16x16x32_bf16(false, a_qv[0], false,
             *(const v16bf*)(rF + rbL * 512 + (size_t)lane * 16), (short)0, cl, false, false);
        cl = __builtin_amdgcn_wmma_f32_16x16x32_bf16(false, a_qv[1], false,
             *(const v16bf*)(rF + (rbL + 1) * 512 + (size_t)lane * 16), (short)0, cl, false, false);
        cr = __builtin_amdgcn_wmma_f32_16x16x32_bf16(false, a_qv[0], false,
             *(const v16bf*)(rF + rbR * 512 + (size_t)lane * 16), (short)0, cr, false, false);
        cr = __builtin_amdgcn_wmma_f32_16x16x32_bf16(false, a_qv[1], false,
             *(const v16bf*)(rF + (rbR + 1) * 512 + (size_t)lane * 16), (short)0, cr, false, false);
        // diagonal gather + mask + scale
#pragma unroll
        for (int r = 0; r < 8; ++r) {
          const int M  = r + 8 * hi;
          const int cp = 15 - M + nI;                     // strip column in [0,30]
          const int srcLane = (cp & 15) | (hi << 4);
          float vL = __shfl(cl[r], srcLane, 32);
          float vR = __shfl(cr[r], srcLane, 32);
          float bd = (cp < 16) ? vL : vR;
          float sc = (ac[r] + bd) * 0.125f;               // 1/sqrt(64)
          if (j0 + nI > i0 + M) sc = -3.0e38f;            // causal mask
          s[sub][r] = sc;
        }
      } else {
#pragma unroll
        for (int r = 0; r < 8; ++r) s[sub][r] = -3.0e38f;
      }
    }
    // ---- online softmax update (per-row stats, 16-lane group reductions) ----
#pragma unroll
    for (int r = 0; r < 8; ++r) {
      float v = fmaxf(s[0][r], s[1][r]);
      for (int off = 1; off < 16; off <<= 1) v = fmaxf(v, __shfl_xor(v, off, 32));
      float mnew  = fmaxf(mrow[r], v);
      float scale = __expf(mrow[r] - mnew);
      float p0 = __expf(s[0][r] - mnew);
      float p1 = __expf(s[1][r] - mnew);
      float rs = p0 + p1;
      for (int off = 1; off < 16; off <<= 1) rs += __shfl_xor(rs, off, 32);
      lrow[r] = lrow[r] * scale + rs;
      mrow[r] = mnew;
      pv[0][r] = p0; pv[1][r] = p1;
      o0[r] *= scale; o1[r] *= scale; o2[r] *= scale; o3[r] *= scale;
    }
    // ---- P: C-layout -> A-layout via row-major LDS tile ----
#pragma unroll
    for (int r = 0; r < 8; ++r) {
      const int M = r + 8 * hi;
      Plds[M][nI]      = (__bf16)pv[0][r];
      Plds[M][16 + nI] = (__bf16)pv[1][r];
    }
    FragU up;                                   // 2 x ds_load_b128
    up.q[0] = *(const uint4*)&Plds[nI][8 * hi];
    up.q[1] = *(const uint4*)&Plds[nI][16 + 8 * hi];
    // ---- O += P @ V (four 16-wide d chunks) ----
    const size_t vb = ((size_t)(b * NH + h) * 64 + js) * 4;
    o0 = __builtin_amdgcn_wmma_f32_16x16x32_bf16(false, up.v, false,
         *(const v16bf*)(vF + (vb + 0) * 512 + (size_t)lane * 16), (short)0, o0, false, false);
    o1 = __builtin_amdgcn_wmma_f32_16x16x32_bf16(false, up.v, false,
         *(const v16bf*)(vF + (vb + 1) * 512 + (size_t)lane * 16), (short)0, o1, false, false);
    o2 = __builtin_amdgcn_wmma_f32_16x16x32_bf16(false, up.v, false,
         *(const v16bf*)(vF + (vb + 2) * 512 + (size_t)lane * 16), (short)0, o2, false, false);
    o3 = __builtin_amdgcn_wmma_f32_16x16x32_bf16(false, up.v, false,
         *(const v16bf*)(vF + (vb + 3) * 512 + (size_t)lane * 16), (short)0, o3, false, false);
  }
  // ---- normalize and store bf16 [b, i, h, d] (row-major [8192, 512]) ----
#pragma unroll
  for (int r = 0; r < 8; ++r) {
    const int M = r + 8 * hi;
    const float inv = 1.0f / lrow[r];
    const size_t row = (size_t)(b * SEQ + i0 + M) * DMOD + h * DH;
    attnB[row +  0 + nI] = (__bf16)(o0[r] * inv);
    attnB[row + 16 + nI] = (__bf16)(o1[r] * inv);
    attnB[row + 32 + nI] = (__bf16)(o2[r] * inv);
    attnB[row + 48 + nI] = (__bf16)(o3[r] * inv);
  }
}

// ---------------------------------------------------------------------------
// Host launcher
// ---------------------------------------------------------------------------
extern "C" void kernel_launch(void* const* d_in, const int* in_sizes, int n_in,
                              void* d_out, int out_size, void* d_ws, size_t ws_size,
                              hipStream_t stream) {
  (void)in_sizes; (void)n_in; (void)out_size; (void)ws_size;
  const float* x   = (const float*)d_in[0];
  const float* pos = (const float*)d_in[1];
  const float* Wq  = (const float*)d_in[2];
  const float* Wk  = (const float*)d_in[3];
  const float* Wv  = (const float*)d_in[4];
  const float* Wr  = (const float*)d_in[5];
  const float* Wo  = (const float*)d_in[6];
  const float* ub  = (const float*)d_in[7];
  const float* vb  = (const float*)d_in[8];

  char* cur = (char*)d_ws;
  auto take = [&](size_t bytes) -> void* {
    void* p = (void*)cur;
    cur += (bytes + 255) & ~(size_t)255;
    return p;
  };

  __bf16* xb  = (__bf16*)take((size_t)ROWS * DMOD * 2);
  __bf16* pb  = (__bf16*)take((size_t)SEQ * DMOD * 2);
  __bf16* wqF = (__bf16*)take((size_t)DMOD * DMOD * 2);
  __bf16* wkF = (__bf16*)take((size_t)DMOD * DMOD * 2);
  __bf16* wvF = (__bf16*)take((size_t)DMOD * DMOD * 2);
  __bf16* wrF = (__bf16*)take((size_t)DMOD * DMOD * 2);
  __bf16* woF = (__bf16*)take((size_t)DMOD * DMOD * 2);
  float*  Qf  = (float*)take((size_t)ROWS * DMOD * 4);
  float*  Kf  = (float*)take((size_t)ROWS * DMOD * 4);
  float*  Vf  = (float*)take((size_t)ROWS * DMOD * 4);
  float*  Rf  = (float*)take((size_t)SEQ * DMOD * 4);
  __bf16* quF = (__bf16*)take((size_t)1 << 23);            // 4M elems * 2B
  __bf16* qvF = (__bf16*)take((size_t)1 << 23);
  __bf16* ktF = (__bf16*)take((size_t)1 << 23);
  __bf16* vF  = (__bf16*)take((size_t)1 << 23);
  __bf16* rF  = (__bf16*)take((size_t)NH * 129 * 2 * 512 * 2);
  __bf16* attnB = (__bf16*)Kf;                             // Kf dead after packing

  auto gemm = [&](const __bf16* A, const __bf16* Bw, float* C, int M) {
    gemm_bf16_kernel<<<M / 16, 256, 0, stream>>>(A, Bw, C, M);
  };

  // 1) activations f32 -> bf16; weights converted+packed to B-frag order
  cvt_bf16_kernel<<<(ROWS * DMOD) / 256, 256, 0, stream>>>(x, xb, ROWS * DMOD);
  cvt_bf16_kernel<<<(SEQ * DMOD) / 256, 256, 0, stream>>>(pos, pb, SEQ * DMOD);
  pack_w_kernel<<<(1 << 18) / 256, 256, 0, stream>>>(Wq, wqF);
  pack_w_kernel<<<(1 << 18) / 256, 256, 0, stream>>>(Wk, wkF);
  pack_w_kernel<<<(1 << 18) / 256, 256, 0, stream>>>(Wv, wvF);
  pack_w_kernel<<<(1 << 18) / 256, 256, 0, stream>>>(Wr, wrF);
  pack_w_kernel<<<(1 << 18) / 256, 256, 0, stream>>>(Wo, woF);

  // 2) projections (WMMA, 16x64 tile per wave, pipelined)
  gemm(xb, wqF, Qf, ROWS);
  gemm(xb, wkF, Kf, ROWS);
  gemm(xb, wvF, Vf, ROWS);
  gemm(pb, wrF, Rf, SEQ);

  // 3) attention-operand fragment packing (bias add fused into qu/qv)
  pack_quqv_kernel<<<(1 << 22) / 256, 256, 0, stream>>>(Qf, ub, vb, quF, qvF);
  pack_k_kernel<<<(1 << 22) / 256, 256, 0, stream>>>(Kf, ktF);
  pack_v_kernel<<<(1 << 22) / 256, 256, 0, stream>>>(Vf, vF);
  pack_r_kernel<<<(NH * 129 * 2 * 512) / 256, 256, 0, stream>>>(Rf, rF);

  // 4) flash attention: one wave per (b, h, 16-row block), bf16 output
  attn_kernel<<<BATCH * NH * (SEQ / 16), 32, 0, stream>>>(quF, qvF, ktF, vF, rF, attnB);

  // 5) output projection (WMMA) -> d_out (f32)
  gemm(attnB, woF, (float*)d_out, ROWS);
}
